// GSA_57509612094191
// MI455X (gfx1250) — matrix operand order repaired and checked by reading, not compiled, not run on hardware
//
#include <hip/hip_runtime.h>
#include <hip/hip_bf16.h>

typedef __bf16 bf16_t;
typedef __attribute__((ext_vector_type(16))) __bf16 v16bf;
typedef __attribute__((ext_vector_type(8)))  __bf16 v8bf;
typedef __attribute__((ext_vector_type(8)))  float  v8f;
typedef __attribute__((ext_vector_type(4)))  unsigned int u32x4;
typedef __attribute__((ext_vector_type(8)))  int i32x8;
typedef __attribute__((ext_vector_type(4)))  int i32x4;

union Frag16 { v16bf v; v8bf h[2]; };

// ---- WMMA fragment helpers (bf16, 16x16x32, wave32) ----

// A fragment: 16(M) x 32(K), rows K-contiguous.
__device__ __forceinline__ v16bf load_frag_a(const bf16_t* base, int ld, int row_base, int k0) {
  int lane = threadIdx.x & 31;
  int l = lane & 15, hi = lane >> 4;
  const bf16_t* p = base + (size_t)(row_base + l) * ld + k0 + hi * 16;
  Frag16 f;
  f.h[0] = *(const v8bf*)(p);
  f.h[1] = *(const v8bf*)(p + 8);
  return f.v;
}

// B fragment: 32(K) x 16(N), stored as B_t rows (N-major, K-contiguous).
__device__ __forceinline__ v16bf load_frag_b(const bf16_t* base, int ld, int col_base, int k0) {
  int lane = threadIdx.x & 31;
  int l = lane & 15, hi = lane >> 4;
  const bf16_t* p = base + (size_t)(col_base + l) * ld + k0 + hi * 16;
  return *(const v16bf*)(p);
}

__device__ __forceinline__ v8f wmma_bf16(v16bf a, v16bf b, v8f c) {
  return __builtin_amdgcn_wmma_f32_16x16x32_bf16(false, a, false, b, (short)0, c, false, false);
}

// Transposed-packed store: out[col][row...] row-contiguous (16B store)
__device__ __forceinline__ void store_packed_bf16(bf16_t* out, int ld, int col_base, int row_base, v8f acc) {
  int lane = threadIdx.x & 31;
  int l = lane & 15, hi = lane >> 4;
  v8bf pk;
#pragma unroll
  for (int i = 0; i < 8; ++i) pk[i] = (bf16_t)acc[i];
  *(v8bf*)(out + (size_t)(col_base + l) * ld + row_base + hi * 8) = pk;
}

__device__ __forceinline__ void store_packed_f32(float* out, int ld, int col_base, int row_base, v8f acc) {
  int lane = threadIdx.x & 31;
  int l = lane & 15, hi = lane >> 4;
  *(v8f*)(out + (size_t)(col_base + l) * ld + row_base + hi * 8) = acc;
}

// Row-major scatter store
__device__ __forceinline__ void store_scatter_bf16(bf16_t* out, int ld, int row_base, int col_base, v8f acc) {
  int lane = threadIdx.x & 31;
  int l = lane & 15, hi = lane >> 4;
#pragma unroll
  for (int i = 0; i < 8; ++i)
    out[(size_t)(row_base + hi * 8 + i) * ld + col_base + l] = (bf16_t)acc[i];
}

// ---- Tensor Data Mover: 2D bf16 tile global->LDS (D# per CDNA5 ISA) ----
__device__ __forceinline__ void tdm_load_2d_bf16(unsigned lds_byte_off, const void* g,
                                                 unsigned width, unsigned rows, unsigned stride) {
  unsigned long long ga = (unsigned long long)(uintptr_t)g;
  u32x4 g0 = { 1u,                                  // count=1
               lds_byte_off,                        // lds_addr (bytes)
               (unsigned)(ga & 0xffffffffu),
               (unsigned)((ga >> 32) & 0x01ffffffu) | 0x80000000u };  // addr[56:32] | type=2<<30
  i32x8 g1 = { (int)(1u << 16),                                        // data_size=1 (2B)
               (int)((width & 0xffffu) << 16),                         // tensor_dim0 lo
               (int)(((width >> 16) & 0xffffu) | ((rows & 0xffffu) << 16)),
               (int)(((rows >> 16) & 0xffffu) | ((width & 0xffffu) << 16)),  // | tile_dim0
               (int)(rows & 0xffffu),                                  // tile_dim1
               (int)stride, 0, 0 };
  i32x4 z4 = { 0, 0, 0, 0 };
#if defined(__clang_major__) && (__clang_major__ >= 23)
  i32x8 z8 = { 0, 0, 0, 0, 0, 0, 0, 0 };
  __builtin_amdgcn_tensor_load_to_lds(g0, g1, z4, z4, z8, 0);
#else
  __builtin_amdgcn_tensor_load_to_lds(g0, g1, z4, z4, 0);
#endif
}

__device__ __forceinline__ unsigned lds_off_of(const void* p) {
  return (unsigned)(uintptr_t)p;
}

// ---- Stage 0: conversions ----

__global__ void k_cvt_bf16(const float* __restrict__ src, bf16_t* __restrict__ dst, int n) {
  int i = blockIdx.x * blockDim.x + threadIdx.x;
  if (i < n) dst[i] = (bf16_t)src[i];
}

// img [b][256][4096] f32 -> imgT [b][4096][256] bf16
__global__ void k_transpose_cvt(const float* __restrict__ img, bf16_t* __restrict__ imgT) {
  __shared__ float tile[32][33];
  int b  = blockIdx.z;
  int n0 = blockIdx.x * 32;
  int c0 = blockIdx.y * 32;
  const float* src = img + ((size_t)b * 256 + c0) * 4096 + n0;
#pragma unroll
  for (int i = 0; i < 4; ++i) {
    int c = threadIdx.y + i * 8;
    tile[c][threadIdx.x] = src[(size_t)c * 4096 + threadIdx.x];
  }
  __syncthreads();
  bf16_t* dst = imgT + ((size_t)b * 4096 + n0) * 256 + c0;
#pragma unroll
  for (int i = 0; i < 4; ++i) {
    int n = threadIdx.y + i * 8;
    dst[(size_t)n * 256 + threadIdx.x] = (bf16_t)tile[threadIdx.x][n];
  }
}

// ---- Stage 1: QKV GEMM  D[n,o] = imgT[b](4096x256) * w_qkv(1536x256)^T ----
__global__ void __launch_bounds__(128) k_qkv(const bf16_t* __restrict__ imgT, const bf16_t* __restrict__ wqkv,
                      bf16_t* __restrict__ qt, bf16_t* __restrict__ kbuf, bf16_t* __restrict__ vbuf) {
  __shared__ bf16_t tB[64 * 256];                 // 32 KB (TDM-staged weights)
  int b    = blockIdx.z;
  int wave = threadIdx.x >> 5;                    // 4 waves
  int mblk = blockIdx.x * 64;
  int n0   = blockIdx.y * 64;                     // o base
  const bf16_t* A = imgT + (size_t)b * 4096 * 256;

  if (wave == 0) {
    tdm_load_2d_bf16(lds_off_of(tB), wqkv + (size_t)n0 * 256, 256, 64, 256);
    __builtin_amdgcn_s_wait_tensorcnt(0);
  }
  __syncthreads();

  int m0 = mblk + wave * 16;
  v8f acc[4] = {};
  for (int k = 0; k < 256; k += 32) {
    v16bf a = load_frag_a(A, 256, m0, k);
#pragma unroll
    for (int t = 0; t < 4; ++t) {
      v16bf bm = load_frag_b(tB, 256, 16 * t, k);   // ds loads
      acc[t] = wmma_bf16(a, bm, acc[t]);
    }
  }
#pragma unroll
  for (int t = 0; t < 4; ++t) {
    int o = n0 + 16 * t;
    if (o < 512) {          // Q -> [b][h][n][64]
      int h = o >> 6, d0 = o & 63;
      bf16_t* dst = qt + ((size_t)b * 8 + h) * 4096 * 64;
      store_scatter_bf16(dst, 64, m0, d0, acc[t]);
    } else if (o < 1024) {  // K -> [b][512][4096]
      bf16_t* dst = kbuf + (size_t)b * 512 * 4096;
      store_packed_bf16(dst, 4096, o - 512, m0, acc[t]);
    } else {                // V
      bf16_t* dst = vbuf + (size_t)b * 512 * 4096;
      store_packed_bf16(dst, 4096, o - 1024, m0, acc[t]);
    }
  }
}

// ---- Stage 2: softmax over n=4096, in place, vectorized 16B accesses ----
__global__ void k_softmax(bf16_t* __restrict__ kbuf) {
  int b = blockIdx.y, r = blockIdx.x;
  bf16_t* row = kbuf + ((size_t)b * 512 + r) * 4096;
  __shared__ float red[8];
  int tid = threadIdx.x;  // 256 threads; each owns 16 contiguous elements
  v8bf u0 = *(const v8bf*)(row + tid * 16);
  v8bf u1 = *(const v8bf*)(row + tid * 16 + 8);
  float vals[16];
  float vmax = -3.4e38f;
#pragma unroll
  for (int i = 0; i < 8; ++i) { vals[i] = (float)u0[i]; vals[8 + i] = (float)u1[i]; }
#pragma unroll
  for (int i = 0; i < 16; ++i) vmax = fmaxf(vmax, vals[i]);
#pragma unroll
  for (int off = 16; off > 0; off >>= 1) vmax = fmaxf(vmax, __shfl_xor(vmax, off, 32));
  if ((tid & 31) == 0) red[tid >> 5] = vmax;
  __syncthreads();
  float m = red[0];
#pragma unroll
  for (int w = 1; w < 8; ++w) m = fmaxf(m, red[w]);
  float s = 0.f;
#pragma unroll
  for (int i = 0; i < 16; ++i) { vals[i] = __expf(vals[i] - m); s += vals[i]; }
#pragma unroll
  for (int off = 16; off > 0; off >>= 1) s += __shfl_xor(s, off, 32);
  __syncthreads();
  if ((tid & 31) == 0) red[tid >> 5] = s;
  __syncthreads();
  float tot = 0.f;
#pragma unroll
  for (int w = 0; w < 8; ++w) tot += red[w];
  float inv = 1.0f / tot;
  v8bf o0, o1;
#pragma unroll
  for (int i = 0; i < 8; ++i) {
    o0[i] = (bf16_t)(vals[i] * inv);
    o1[i] = (bf16_t)(vals[8 + i] * inv);
  }
  *(v8bf*)(row + tid * 16)     = o0;
  *(v8bf*)(row + tid * 16 + 8) = o1;
}

// ---- Stage 3a: split-K partial context. part[p][b*8+h][e][d] f32, K-chunk p*1024 ----
__global__ void __launch_bounds__(256) k_context(const bf16_t* __restrict__ kbuf, const bf16_t* __restrict__ vbuf,
                          float* __restrict__ ctx_part) {
  int b = blockIdx.y, h = blockIdx.x;
  int ks = blockIdx.z * 1024;
  int wave = threadIdx.x >> 5;        // 8 waves
  int d0 = (wave & 3) * 16;
  int e0 = (wave >> 2) * 32;
  const bf16_t* A = kbuf + ((size_t)b * 512 + h * 64) * 4096;
  const bf16_t* B = vbuf + ((size_t)b * 512 + h * 64) * 4096;
  v8f acc0 = {}, acc1 = {};
  v16bf a  = load_frag_a(A, 4096, d0, ks);
  v16bf b0 = load_frag_b(B, 4096, e0, ks);
  v16bf b1 = load_frag_b(B, 4096, e0 + 16, ks);
  for (int k = ks; k < ks + 1024 - 32; k += 32) {
    v16bf an  = load_frag_a(A, 4096, d0, k + 32);
    v16bf b0n = load_frag_b(B, 4096, e0, k + 32);
    v16bf b1n = load_frag_b(B, 4096, e0 + 16, k + 32);
    acc0 = wmma_bf16(a, b0, acc0);
    acc1 = wmma_bf16(a, b1, acc1);
    a = an; b0 = b0n; b1 = b1n;
  }
  acc0 = wmma_bf16(a, b0, acc0);
  acc1 = wmma_bf16(a, b1, acc1);
  float* out = ctx_part + ((size_t)blockIdx.z * 64 + b * 8 + h) * 4096;  // [e][d] f32
  store_packed_f32(out, 64, e0, d0, acc0);
  store_packed_f32(out, 64, e0 + 16, d0, acc1);
}

// ---- Stage 3b: reduce 4 partials -> ctxT bf16 ----
__global__ void k_ctx_reduce(const float* __restrict__ part, bf16_t* __restrict__ ctxT) {
  int i = blockIdx.x * 256 + threadIdx.x;           // over 64*4096 = 262144
  float s = part[i] + part[i + 262144] + part[i + 2 * 262144] + part[i + 3 * 262144];
  ctxT[i] = (bf16_t)s;
}

// ---- Stage 4: content^T[n,e] = sum_d qT[n,d]*ctxT[e,d]; store contT[b][n][h*64+e] ----
__global__ void __launch_bounds__(128) k_content(const bf16_t* __restrict__ qt, const bf16_t* __restrict__ ctxT,
                          bf16_t* __restrict__ contT) {
  int b = blockIdx.z, h = blockIdx.y;
  int wave = threadIdx.x >> 5;            // 4 waves
  int m0 = blockIdx.x * 64 + wave * 16;
  const bf16_t* A = qt   + ((size_t)b * 8 + h) * 4096 * 64;  // [n][64]
  const bf16_t* B = ctxT + ((size_t)b * 8 + h) * 64 * 64;    // [e][64]
  v8f acc[4] = {};
  for (int k = 0; k < 64; k += 32) {
    v16bf a = load_frag_a(A, 64, m0, k);
#pragma unroll
    for (int t = 0; t < 4; ++t) {
      v16bf bm = load_frag_b(B, 64, 16 * t, k);
      acc[t] = wmma_bf16(a, bm, acc[t]);
    }
  }
  bf16_t* out = contT + (size_t)b * 4096 * 512 + h * 64;     // [n][512]
#pragma unroll
  for (int t = 0; t < 4; ++t) store_scatter_bf16(out, 512, m0, 16 * t, acc[t]);
}

// ---- Stage 5: out[b][o][n] = sum_c w_out[o,c]*contT[n,c] + b_out[o] ----
// Double-buffered TDM: both 32KB weight halves issued up front; compute on half0
// overlaps the in-flight load of half1 (partial s_wait_tensorcnt).
__global__ void __launch_bounds__(128) k_out(const bf16_t* __restrict__ contT, const bf16_t* __restrict__ wout,
                      const float* __restrict__ bout, float* __restrict__ out) {
  __shared__ bf16_t tB0[64 * 256];                // 32 KB
  __shared__ bf16_t tB1[64 * 256];                // 32 KB
  int b = blockIdx.z;
  int wave = threadIdx.x >> 5;                    // 4 waves
  int m0 = blockIdx.x * 64 + wave * 16;           // n base
  int n0 = blockIdx.y * 64;                       // o base
  const bf16_t* A = contT + (size_t)b * 4096 * 512;
  v8f acc[4] = {};

  if (wave == 0) {
    tdm_load_2d_bf16(lds_off_of(tB0), wout + (size_t)n0 * 512,       256, 64, 512);
    tdm_load_2d_bf16(lds_off_of(tB1), wout + (size_t)n0 * 512 + 256, 256, 64, 512);
    __builtin_amdgcn_s_wait_tensorcnt(1);         // half0 landed; half1 may be in flight
  }
  __syncthreads();
  for (int k = 0; k < 256; k += 32) {
    v16bf a = load_frag_a(A, 512, m0, k);
#pragma unroll
    for (int t = 0; t < 4; ++t) acc[t] = wmma_bf16(a, load_frag_b(tB0, 256, 16 * t, k), acc[t]);
  }
  if (wave == 0) __builtin_amdgcn_s_wait_tensorcnt(0);
  __syncthreads();
  for (int k = 0; k < 256; k += 32) {
    v16bf a = load_frag_a(A, 512, m0, 256 + k);
#pragma unroll
    for (int t = 0; t < 4; ++t) acc[t] = wmma_bf16(a, load_frag_b(tB1, 256, 16 * t, k), acc[t]);
  }

  int lane = threadIdx.x & 31;
  int l = lane & 15, hi = lane >> 4;
#pragma unroll
  for (int t = 0; t < 4; ++t) {
    int o = n0 + 16 * t + l;
    float bb = bout[o];
    float* dst = out + ((size_t)b * 256 + o) * 4096 + m0 + hi * 8;
    v8f r;
#pragma unroll
    for (int i = 0; i < 8; ++i) r[i] = acc[t][i] + bb;
    *(v8f*)dst = r;
  }
}

extern "C" void kernel_launch(void* const* d_in, const int* in_sizes, int n_in,
                              void* d_out, int out_size, void* d_ws, size_t ws_size,
                              hipStream_t stream) {
  const float* img  = (const float*)d_in[0];   // [8,256,64,64]
  const float* wqkv = (const float*)d_in[1];   // [1536,256]
  const float* wout = (const float*)d_in[2];   // [256,512]
  const float* bout = (const float*)d_in[3];   // [256]
  float* out = (float*)d_out;

  char* ws = (char*)d_ws;
  size_t off = 0;
  bf16_t* imgT  = (bf16_t*)(ws + off); off += (size_t)8 * 4096 * 256 * 2;
  bf16_t* wqkvb = (bf16_t*)(ws + off); off += (size_t)1536 * 256 * 2;
  bf16_t* woutb = (bf16_t*)(ws + off); off += (size_t)256 * 512 * 2;
  bf16_t* qt    = (bf16_t*)(ws + off); off += (size_t)8 * 512 * 4096 * 2;
  bf16_t* kb    = (bf16_t*)(ws + off); off += (size_t)8 * 512 * 4096 * 2;
  bf16_t* vb    = (bf16_t*)(ws + off); off += (size_t)8 * 512 * 4096 * 2;
  bf16_t* ctxT  = (bf16_t*)(ws + off); off += (size_t)8 * 8 * 64 * 64 * 2;
  bf16_t* contT = (bf16_t*)(ws + off); off += (size_t)8 * 4096 * 512 * 2;
  float*  ctxP  = (float*) (ws + off); off += (size_t)4 * 64 * 4096 * 4;   // split-K partials

  k_cvt_bf16<<<dim3((1536 * 256) / 256), 256, 0, stream>>>(wqkv, wqkvb, 1536 * 256);
  k_cvt_bf16<<<dim3((256 * 512) / 256), 256, 0, stream>>>(wout, woutb, 256 * 512);
  k_transpose_cvt<<<dim3(128, 8, 8), dim3(32, 8), 0, stream>>>(img, imgT);
  k_qkv<<<dim3(64, 24, 8), 128, 0, stream>>>(imgT, wqkvb, qt, kb, vb);
  k_softmax<<<dim3(512, 8), 256, 0, stream>>>(kb);
  k_context<<<dim3(8, 8, 4), 256, 0, stream>>>(kb, vb, ctxP);
  k_ctx_reduce<<<dim3(262144 / 256), 256, 0, stream>>>(ctxP, ctxT);
  k_content<<<dim3(64, 8, 8), 128, 0, stream>>>(qt, ctxT, contT);
  k_out<<<dim3(64, 4, 8), 128, 0, stream>>>(contT, woutb, bout, out);
}
